// SynProtXgatfp3_predictor_36223754174561
// MI455X (gfx1250) — matrix operand order repaired and checked by reading, not compiled, not run on hardware
//
#include <hip/hip_runtime.h>
#include <math.h>

typedef __attribute__((ext_vector_type(16))) _Float16 v16h;
typedef __attribute__((ext_vector_type(8)))  float    v8f;

namespace {

constexpr int BM  = 128;   // M tile (all GEMM M dims are multiples of 128)
constexpr int BN  = 64;    // N tile
constexpr int BKT = 32;    // K tile == WMMA K
constexpr int LDP = 18;    // padded pair (2xf16 = dword) stride: 16 pairs + 2 pad

union PairU { _Float16 h[2]; unsigned u; };
union FragU { unsigned u[8]; v16h v; };

// one LDS->frag->WMMA stage over the currently staged BKT slice
__device__ __forceinline__ void wmma_stage(const unsigned (&AsU)[BM][LDP],
                                           const unsigned (&BsU)[BN][LDP],
                                           v8f (&acc)[2][2],
                                           int lane, int waveM, int waveN) {
  // Fragment loads per CDNA5 16-bit WMMA layouts.
  // A 16x32: lanes 0-15 -> K 0..7 (pairs kb..kb+3) & 16..23; lanes 16-31 -> K 8..15 & 24..31
  // B 32x16: same K order as A: lanes 0-15 -> K 0..7 & 16..23; lanes 16-31 -> K 8..15 & 24..31
  FragU af[2], bf[2];
  const int lrow = lane & 15;
  const int kbA  = (lane >> 4) * 4;
  const int pbB  = (lane >> 4) * 4;
  #pragma unroll
  for (int mi = 0; mi < 2; mi++) {
    const int r = waveM * 32 + mi * 16 + lrow;
    #pragma unroll
    for (int j = 0; j < 4; j++) {
      af[mi].u[j]     = AsU[r][kbA + j];
      af[mi].u[j + 4] = AsU[r][8 + kbA + j];
    }
  }
  #pragma unroll
  for (int ni = 0; ni < 2; ni++) {
    const int c = waveN * 32 + ni * 16 + lrow;
    #pragma unroll
    for (int j = 0; j < 8; j++) bf[ni].u[j] = BsU[c][pbB + (j & 3) + (j >> 2) * 8];
  }
  #pragma unroll
  for (int mi = 0; mi < 2; mi++)
    #pragma unroll
    for (int ni = 0; ni < 2; ni++)
      acc[mi][ni] = __builtin_amdgcn_wmma_f32_16x16x32_f16(
          false, af[mi].v, false, bf[ni].v, (short)0, acc[mi][ni], false, false);
}

// ---------------------------------------------------------------------------
// Tiled WMMA GEMM:  C[M,N] = act(A[M,K] (*rowScale) @ W[K,N] + bias)
// ACT: 0 = none, 1 = relu, 2 = elu.   REQUIRES: M % 128 == 0.
// Grid: x = M tiles (fast dim -> co-resident blocks share weight stream in L2),
//       y = N tiles.
// ---------------------------------------------------------------------------
template<int ACT>
__global__ __launch_bounds__(256)
void gemm_wmma_kernel(const float* __restrict__ A, const float* __restrict__ W,
                      const float* __restrict__ bias, float* __restrict__ Cmat,
                      int M, int N, int K, const float* __restrict__ rowScale)
{
  __shared__ unsigned AsU[BM][LDP];   // A tile, f16 pairs, K-contiguous
  __shared__ unsigned BsU[BN][LDP];   // B tile transposed, f16 pairs, K-contiguous

  const int tid   = threadIdx.x;
  const int lane  = tid & 31;
  const int wave  = tid >> 5;
  const int waveM = wave & 3;   // 4 waves along M (32 rows each)
  const int waveN = wave >> 2;  // 2 waves along N (32 cols each)
  const int m0 = blockIdx.x * BM;
  const int n0 = blockIdx.y * BN;

  v8f zero = {0.f,0.f,0.f,0.f,0.f,0.f,0.f,0.f};
  v8f acc[2][2];
  acc[0][0] = zero; acc[0][1] = zero; acc[1][0] = zero; acc[1][1] = zero;

  // A-fill coordinates: 128 rows x 16 pairs, 8 pairs (16 consecutive k) per thread
  const int aRow      = tid >> 1;
  const int aPairBase = (tid & 1) * 8;
  const float aScale  = rowScale ? rowScale[m0 + aRow] : 1.f;  // row always < M
  const float* __restrict__ arow = A + (size_t)(m0 + aRow) * K;

  // B-fill coordinates: 64 cols x 8 consecutive k each, n-coalesced.
  // Column guard is uniform per thread: clamp address once, fold 0/1 mask into convert.
  const int   bN     = tid & 63;
  const int   bGroup = tid >> 6;              // 0..3 -> k sub-range of 8
  const int   gnc    = (n0 + bN) < N ? (n0 + bN) : (N - 1);
  const float nMask  = (n0 + bN) < N ? 1.f : 0.f;

  const int kFull = K & ~(BKT - 1);

  float aReg[16];  // next A slice (16 consecutive k)
  float bReg[8];   // next B slice (8 strided k)

  auto load_stage = [&](int k0) {
    const float* __restrict__ ap = arow + k0 + aPairBase * 2;
    #pragma unroll
    for (int q = 0; q < 16; q++) aReg[q] = ap[q];          // -> global_load_b128 x4
    const float* __restrict__ wq = W + (size_t)(k0 + bGroup * 8) * N + gnc;
    #pragma unroll
    for (int q = 0; q < 8; q++) bReg[q] = wq[(size_t)q * N]; // 8 independent loads, 1 wait
  };
  auto store_stage = [&]() {
    #pragma unroll
    for (int j = 0; j < 8; j++) {
      PairU p;
      p.h[0] = (_Float16)(aReg[2*j]     * aScale);
      p.h[1] = (_Float16)(aReg[2*j + 1] * aScale);
      AsU[aRow][aPairBase + j] = p.u;
    }
    #pragma unroll
    for (int j = 0; j < 4; j++) {
      PairU p;
      p.h[0] = (_Float16)(bReg[2*j]     * nMask);
      p.h[1] = (_Float16)(bReg[2*j + 1] * nMask);
      BsU[bN][bGroup * 4 + j] = p.u;
    }
  };

  if (kFull > 0) load_stage(0);
  for (int k0 = 0; k0 < kFull; k0 += BKT) {
    store_stage();
    __syncthreads();
    if (k0 + BKT < kFull) load_stage(k0 + BKT);  // overlap next-tile loads with WMMA
    wmma_stage(AsU, BsU, acc, lane, waveM, waveN);
    __syncthreads();
  }

  if (kFull < K) {  // single guarded tail stage: clamped addresses + cndmask zero-select
    {
      #pragma unroll
      for (int j = 0; j < 8; j++) {
        const int pr = aPairBase + j;
        const int k  = kFull + pr * 2;
        const int kc0 = k     < K ? k     : (K - 1);
        const int kc1 = k + 1 < K ? k + 1 : (K - 1);
        float f0 = arow[kc0] * aScale; f0 = k     < K ? f0 : 0.f;
        float f1 = arow[kc1] * aScale; f1 = k + 1 < K ? f1 : 0.f;
        PairU p; p.h[0] = (_Float16)f0; p.h[1] = (_Float16)f1;
        AsU[aRow][pr] = p.u;
      }
    }
    {
      #pragma unroll
      for (int j = 0; j < 4; j++) {
        const int pr = bGroup * 4 + j;
        const int k  = kFull + pr * 2;
        const int kc0 = k     < K ? k     : (K - 1);
        const int kc1 = k + 1 < K ? k + 1 : (K - 1);
        float f0 = W[(size_t)kc0 * N + gnc] * nMask; f0 = k     < K ? f0 : 0.f;
        float f1 = W[(size_t)kc1 * N + gnc] * nMask; f1 = k + 1 < K ? f1 : 0.f;
        PairU p; p.h[0] = (_Float16)f0; p.h[1] = (_Float16)f1;
        BsU[bN][pr] = p.u;
      }
    }
    __syncthreads();
    wmma_stage(AsU, BsU, acc, lane, waveM, waveN);
    __syncthreads();
  }

  // Epilogue: C/D 16x16 f32 layout: lane -> col, vgpr v -> row (0..7 / 8..15)
  const int lrow = lane & 15;
  const int rsel = (lane >> 4) * 8;
  #pragma unroll
  for (int mi = 0; mi < 2; mi++) {
    #pragma unroll
    for (int ni = 0; ni < 2; ni++) {
      const int col     = n0 + waveN * 32 + ni * 16 + lrow;
      const int rowBase = m0 + waveM * 32 + mi * 16 + rsel;
      if (col < N) {
        const float bv = bias ? bias[col] : 0.f;
        #pragma unroll
        for (int v = 0; v < 8; v++) {
          const int row = rowBase + v;  // always < M (M % 128 == 0)
          float x = acc[mi][ni][v] + bv;
          if (ACT == 1) x = fmaxf(x, 0.f);
          else if (ACT == 2) x = x > 0.f ? x : (expf(x) - 1.f);
          Cmat[(size_t)row * N + col] = x;
        }
      }
    }
  }
}

// ---------------------------------------------------------------------------
// GAT helpers
// ---------------------------------------------------------------------------
__global__ void fill_kernel(float* __restrict__ p, float v, size_t n) {
  size_t i = (size_t)blockIdx.x * blockDim.x + threadIdx.x;
  if (i < n) p[i] = v;
}

__device__ inline void atomicMaxFloat(float* addr, float val) {
  int iv = __float_as_int(val);
  if (iv >= 0) atomicMax((int*)addr, iv);
  else         atomicMin((unsigned int*)addr, (unsigned int)iv);
}

// per (node, head): dot(h, a_src) and dot(h, a_dst)
__global__ void node_att_kernel(const float* __restrict__ h,
                                const float* __restrict__ a_src,
                                const float* __restrict__ a_dst,
                                float* __restrict__ as_, float* __restrict__ ad_,
                                int Ntot, int Hn, int Cc) {
  int idx = blockIdx.x * blockDim.x + threadIdx.x;
  if (idx >= Ntot * Hn) return;
  int n = idx / Hn, hh = idx % Hn;
  const float* hp  = h + ((size_t)n * Hn + hh) * Cc;
  const float* sv  = a_src + hh * Cc;
  const float* dv  = a_dst + hh * Cc;
  float s = 0.f, d = 0.f;
  for (int c = 0; c < Cc; c++) { float v = hp[c]; s += v * sv[c]; d += v * dv[c]; }
  as_[idx] = s; ad_[idx] = d;
}

__device__ inline void edge_ends(const int* ei, int e, int E, int& s, int& d) {
  if (e < E) { s = ei[e]; d = ei[E + e]; } else { s = d = e - E; }  // self-loops appended
}

__global__ void edge_max_kernel(const int* __restrict__ ei,
                                const float* __restrict__ as_, const float* __restrict__ ad_,
                                float* __restrict__ m, int E, int Ntot, int Hn) {
  int idx = blockIdx.x * blockDim.x + threadIdx.x;
  int EP = E + Ntot;
  if (idx >= EP * Hn) return;
  int e = idx / Hn, hh = idx % Hn, s, d;
  edge_ends(ei, e, E, s, d);
  float lg = as_[s * Hn + hh] + ad_[d * Hn + hh];
  lg = lg > 0.f ? lg : 0.2f * lg;                 // leaky_relu(0.2)
  atomicMaxFloat(&m[d * Hn + hh], lg);
}

__global__ void edge_sum_kernel(const int* __restrict__ ei,
                                const float* __restrict__ as_, const float* __restrict__ ad_,
                                const float* __restrict__ m, float* __restrict__ den,
                                int E, int Ntot, int Hn) {
  int idx = blockIdx.x * blockDim.x + threadIdx.x;
  int EP = E + Ntot;
  if (idx >= EP * Hn) return;
  int e = idx / Hn, hh = idx % Hn, s, d;
  edge_ends(ei, e, E, s, d);
  float lg = as_[s * Hn + hh] + ad_[d * Hn + hh];
  lg = lg > 0.f ? lg : 0.2f * lg;
  atomicAdd(&den[d * Hn + hh], expf(lg - m[d * Hn + hh]));
}

// block = Cc channels, grid = E' edges; loops heads
__global__ void edge_accum_kernel(const int* __restrict__ ei,
                                  const float* __restrict__ as_, const float* __restrict__ ad_,
                                  const float* __restrict__ m, const float* __restrict__ den,
                                  const float* __restrict__ h, float* __restrict__ out,
                                  int E, int Ntot, int Hn, int Cc) {
  int e = blockIdx.x, s, d;
  edge_ends(ei, e, E, s, d);
  int c = threadIdx.x;
  for (int hh = 0; hh < Hn; hh++) {
    float lg = as_[s * Hn + hh] + ad_[d * Hn + hh];
    lg = lg > 0.f ? lg : 0.2f * lg;
    float alpha = expf(lg - m[d * Hn + hh]) / (den[d * Hn + hh] + 1e-16f);
    atomicAdd(&out[((size_t)d * Hn + hh) * Cc + c],
              h[((size_t)s * Hn + hh) * Cc + c] * alpha);
  }
}

__global__ void bias_elu_kernel(float* __restrict__ x, const float* __restrict__ b,
                                size_t rows, int cols) {
  size_t i = (size_t)blockIdx.x * blockDim.x + threadIdx.x;
  if (i >= rows * (size_t)cols) return;
  int c = (int)(i % cols);
  float v = x[i] + b[c];
  x[i] = v > 0.f ? v : (expf(v) - 1.f);
}

// global max pool over npg consecutive nodes; block = Cc threads, grid = B graphs
__global__ void pool_kernel(const float* __restrict__ x, float* __restrict__ out,
                            int npg, int Cc) {
  int b = blockIdx.x, c = threadIdx.x;
  float mx = -INFINITY;
  for (int n = 0; n < npg; n++)
    mx = fmaxf(mx, x[((size_t)b * npg + n) * Cc + c]);
  out[(size_t)b * Cc + c] = mx;
}

// per-row inverse L2 norm (for gene branch, fused into GEMM A-load)
__global__ void rownorm_kernel(const float* __restrict__ x, float* __restrict__ scale, int cols) {
  __shared__ float red[256];
  int b = blockIdx.x;
  const float* row = x + (size_t)b * cols;
  float s = 0.f;
  for (int c = threadIdx.x; c < cols; c += 256) { float v = row[c]; s += v * v; }
  red[threadIdx.x] = s; __syncthreads();
  for (int off = 128; off > 0; off >>= 1) {
    if (threadIdx.x < off) red[threadIdx.x] += red[threadIdx.x + off];
    __syncthreads();
  }
  if (threadIdx.x == 0) scale[b] = 1.f / fmaxf(sqrtf(red[0]), 1e-12f);
}

// copy src[rows,scols] into dst[rows,dcols] at column offset doff
__global__ void copy_cols_kernel(const float* __restrict__ src, float* __restrict__ dst,
                                 int rows, int scols, int dcols, int doff) {
  size_t i = (size_t)blockIdx.x * blockDim.x + threadIdx.x;
  if (i >= (size_t)rows * scols) return;
  int r = (int)(i / scols), c = (int)(i % scols);
  dst[(size_t)r * dcols + doff + c] = src[i];
}

inline void launch_gemm(hipStream_t st, const float* A, const float* W, const float* bias,
                        float* C, int M, int N, int K, int act, const float* rs = nullptr) {
  dim3 g((M + BM - 1) / BM, (N + BN - 1) / BN), b(256);  // M tiles fastest
  if (act == 0)      gemm_wmma_kernel<0><<<g, b, 0, st>>>(A, W, bias, C, M, N, K, rs);
  else if (act == 1) gemm_wmma_kernel<1><<<g, b, 0, st>>>(A, W, bias, C, M, N, K, rs);
  else               gemm_wmma_kernel<2><<<g, b, 0, st>>>(A, W, bias, C, M, N, K, rs);
}

inline dim3 cdiv256(size_t n) { return dim3((unsigned)((n + 255) / 256)); }

} // namespace

// ---------------------------------------------------------------------------
extern "C" void kernel_launch(void* const* d_in, const int* in_sizes, int n_in,
                              void* d_out, int out_size, void* d_ws, size_t ws_size,
                              hipStream_t stream) {
  (void)n_in; (void)out_size; (void)ws_size;

  // ---- inputs (setup_inputs dict order) ----
  const float* x1       = (const float*)d_in[0];
  const float* x2       = (const float*)d_in[1];
  const float* gene_exp = (const float*)d_in[2];
  const float* drr      = (const float*)d_in[3];
  const float* drc      = (const float*)d_in[4];
  const float* prot_exp = (const float*)d_in[5];
  const int*   ei1      = (const int*)d_in[6];
  const int*   ei2      = (const int*)d_in[7];
  // d_in[8] = batch (nodes are contiguous per graph; structure used directly)
  const float* W_gat1 = (const float*)d_in[9],  *b_gat1 = (const float*)d_in[10];
  const float* a_src1 = (const float*)d_in[11], *a_dst1 = (const float*)d_in[12];
  const float* W_gat2 = (const float*)d_in[13], *b_gat2 = (const float*)d_in[14];
  const float* a_src2 = (const float*)d_in[15], *a_dst2 = (const float*)d_in[16];
  const float* W_fcg  = (const float*)d_in[17], *b_fcg  = (const float*)d_in[18];
  const float* Wr1 = (const float*)d_in[19], *br1 = (const float*)d_in[20];
  const float* Wr2 = (const float*)d_in[21], *br2 = (const float*)d_in[22];
  const float* Wr3 = (const float*)d_in[23], *br3 = (const float*)d_in[24];
  const float* Wp1 = (const float*)d_in[25], *bp1 = (const float*)d_in[26];
  const float* Wp2 = (const float*)d_in[27], *bp2 = (const float*)d_in[28];
  const float* Wp3 = (const float*)d_in[29], *bp3 = (const float*)d_in[30];
  const float* Wdrr = (const float*)d_in[31], *bdrr = (const float*)d_in[32];
  const float* Wdrc = (const float*)d_in[33], *bdrc = (const float*)d_in[34];
  const float* Wf1 = (const float*)d_in[35], *bf1 = (const float*)d_in[36];
  const float* Wf2 = (const float*)d_in[37], *bf2 = (const float*)d_in[38];
  const float* Wout = (const float*)d_in[39], *bout = (const float*)d_in[40];

  // ---- dimensions ----
  const int IN_FEAT = 78, Cch = 128, Hh = 8, HC = Hh * Cch; // 1024
  const int GENE = 18000, PROT = 6688, FP = 2048;
  const int NT  = in_sizes[8];            // total nodes (32768)
  const int Bn  = in_sizes[2] / GENE;     // graphs (1024)
  const int NPG = NT / Bn;                // nodes/graph (32)
  const int E   = in_sizes[6] / 2;        // edges/branch (131072)

  // ---- workspace carve ----
  float* w = (float*)d_ws;
  size_t off = 0;
  auto alloc = [&](size_t n) { float* p = w + off; off += n; return p; };
  float* bufH  = alloc((size_t)NT * HC);   // node features h (pre-attention)
  float* bufO  = alloc((size_t)NT * HC);   // GAT output / next-layer input
  float* asb   = alloc((size_t)NT * Hh);
  float* adb   = alloc((size_t)NT * Hh);
  float* mb    = alloc((size_t)NT * Hh);
  float* denb  = alloc((size_t)NT * Hh);
  float* gp    = alloc((size_t)Bn * Cch);  // pooled graph embedding (pre-fc)
  float* g1    = alloc((size_t)Bn * Cch);
  float* g2    = alloc((size_t)Bn * Cch);
  float* scale = alloc((size_t)Bn);
  float* c1    = alloc((size_t)Bn * 2048);
  float* c2    = alloc((size_t)Bn * 512);
  float* cellv = alloc((size_t)Bn * 256);
  float* p1    = alloc((size_t)Bn * 1024);
  float* p2    = alloc((size_t)Bn * 512);
  float* p3    = alloc((size_t)Bn * 256);
  float* cat   = alloc((size_t)Bn * (Cch + FP));   // 2176
  float* d1    = alloc((size_t)Bn * (Cch + FP));
  float* d2    = alloc((size_t)Bn * (Cch + FP));
  float* ycat  = alloc((size_t)Bn * 4864);
  float* f1b   = alloc((size_t)Bn * 512);
  float* f2b   = alloc((size_t)Bn * 256);

  const int EP = E + NT;  // edges + self-loops

  // ---- drug branch (shared buffers, run twice) ----
  auto drug_branch = [&](const float* x, const int* ei, float* gout) {
    // GAT layer 1: h = x @ W_gat1   [NT x 1024]
    launch_gemm(stream, x, W_gat1, nullptr, bufH, NT, HC, IN_FEAT, 0);
    node_att_kernel<<<cdiv256((size_t)NT * Hh), 256, 0, stream>>>(
        bufH, a_src1, a_dst1, asb, adb, NT, Hh, Cch);
    fill_kernel<<<cdiv256((size_t)NT * Hh), 256, 0, stream>>>(mb, -INFINITY, (size_t)NT * Hh);
    fill_kernel<<<cdiv256((size_t)NT * Hh), 256, 0, stream>>>(denb, 0.f, (size_t)NT * Hh);
    fill_kernel<<<cdiv256((size_t)NT * HC), 256, 0, stream>>>(bufO, 0.f, (size_t)NT * HC);
    edge_max_kernel<<<cdiv256((size_t)EP * Hh), 256, 0, stream>>>(ei, asb, adb, mb, E, NT, Hh);
    edge_sum_kernel<<<cdiv256((size_t)EP * Hh), 256, 0, stream>>>(ei, asb, adb, mb, denb, E, NT, Hh);
    edge_accum_kernel<<<dim3(EP), dim3(Cch), 0, stream>>>(ei, asb, adb, mb, denb, bufH, bufO, E, NT, Hh, Cch);
    bias_elu_kernel<<<cdiv256((size_t)NT * HC), 256, 0, stream>>>(bufO, b_gat1, NT, HC);

    // GAT layer 2: h2 = x' @ W_gat2   [NT x 128], 1 head
    launch_gemm(stream, bufO, W_gat2, nullptr, bufH, NT, Cch, HC, 0);
    node_att_kernel<<<cdiv256((size_t)NT), 256, 0, stream>>>(
        bufH, a_src2, a_dst2, asb, adb, NT, 1, Cch);
    fill_kernel<<<cdiv256((size_t)NT), 256, 0, stream>>>(mb, -INFINITY, (size_t)NT);
    fill_kernel<<<cdiv256((size_t)NT), 256, 0, stream>>>(denb, 0.f, (size_t)NT);
    fill_kernel<<<cdiv256((size_t)NT * Cch), 256, 0, stream>>>(bufO, 0.f, (size_t)NT * Cch);
    edge_max_kernel<<<cdiv256((size_t)EP), 256, 0, stream>>>(ei, asb, adb, mb, E, NT, 1);
    edge_sum_kernel<<<cdiv256((size_t)EP), 256, 0, stream>>>(ei, asb, adb, mb, denb, E, NT, 1);
    edge_accum_kernel<<<dim3(EP), dim3(Cch), 0, stream>>>(ei, asb, adb, mb, denb, bufH, bufO, E, NT, 1, Cch);
    bias_elu_kernel<<<cdiv256((size_t)NT * Cch), 256, 0, stream>>>(bufO, b_gat2, NT, Cch);

    // global max pool + fc
    pool_kernel<<<dim3(Bn), dim3(Cch), 0, stream>>>(bufO, gp, NPG, Cch);
    launch_gemm(stream, gp, W_fcg, b_fcg, gout, Bn, Cch, Cch, 1);
  };
  drug_branch(x1, ei1, g1);
  drug_branch(x2, ei2, g2);

  // ---- gene branch: L2-normalize fused into Wr1 GEMM ----
  rownorm_kernel<<<dim3(Bn), dim3(256), 0, stream>>>(gene_exp, scale, GENE);
  launch_gemm(stream, gene_exp, Wr1, br1, c1, Bn, 2048, GENE, 1, scale);
  launch_gemm(stream, c1, Wr2, br2, c2, Bn, 512, 2048, 1);
  launch_gemm(stream, c2, Wr3, br3, cellv, Bn, 256, 512, 1);

  // ---- protein branch ----
  launch_gemm(stream, prot_exp, Wp1, bp1, p1, Bn, 1024, PROT, 2);
  launch_gemm(stream, p1, Wp2, bp2, p2, Bn, 512, 1024, 2);
  launch_gemm(stream, p2, Wp3, bp3, p3, Bn, 256, 512, 1);

  // ---- fingerprint fusion ----
  const int CF = Cch + FP; // 2176
  copy_cols_kernel<<<cdiv256((size_t)Bn * Cch), 256, 0, stream>>>(g1, cat, Bn, Cch, CF, 0);
  copy_cols_kernel<<<cdiv256((size_t)Bn * FP), 256, 0, stream>>>(drr, cat, Bn, FP, CF, Cch);
  launch_gemm(stream, cat, Wdrr, bdrr, d1, Bn, CF, CF, 1);
  copy_cols_kernel<<<cdiv256((size_t)Bn * Cch), 256, 0, stream>>>(g2, cat, Bn, Cch, CF, 0);
  copy_cols_kernel<<<cdiv256((size_t)Bn * FP), 256, 0, stream>>>(drc, cat, Bn, FP, CF, Cch);
  launch_gemm(stream, cat, Wdrc, bdrc, d2, Bn, CF, CF, 1);

  // ---- predictor ----
  const int YD = 2 * CF + 512; // 4864
  copy_cols_kernel<<<cdiv256((size_t)Bn * CF), 256, 0, stream>>>(d1, ycat, Bn, CF, YD, 0);
  copy_cols_kernel<<<cdiv256((size_t)Bn * CF), 256, 0, stream>>>(d2, ycat, Bn, CF, YD, CF);
  copy_cols_kernel<<<cdiv256((size_t)Bn * 256), 256, 0, stream>>>(cellv, ycat, Bn, 256, YD, 2 * CF);
  copy_cols_kernel<<<cdiv256((size_t)Bn * 256), 256, 0, stream>>>(p3, ycat, Bn, 256, YD, 2 * CF + 256);
  launch_gemm(stream, ycat, Wf1, bf1, f1b, Bn, 512, YD, 1);
  launch_gemm(stream, f1b, Wf2, bf2, f2b, Bn, 256, 512, 1);
  launch_gemm(stream, f2b, Wout, bout, (float*)d_out, Bn, 1, 256, 0);
}
